// SparseMiddleExtractor_15719580304500
// MI455X (gfx1250) — matrix operand order repaired and checked by reading, not compiled, not run on hardware
//
#include <hip/hip_runtime.h>
#include <hip/hip_bf16.h>

// ---------------------------------------------------------------------------
// SparseMiddleExtractor on gfx1250 via WMMA implicit-GEMM convolutions.
// f16 activations/weights, f32 WMMA accumulation (v_wmma_f32_16x16x32_f16).
// Conv waves: 2 y-rows x 16 x-positions (M) x 64 out-channels (4 N-tiles)
// -> 8 WMMAs per B-fragment set. Interior tiles run a fully branchless
// kernel (constant-folded load offsets); boundary tiles run an edge kernel.
// ---------------------------------------------------------------------------

typedef __attribute__((ext_vector_type(16))) _Float16 v16h;
typedef __attribute__((ext_vector_type(8)))  _Float16 v8h;
typedef __attribute__((ext_vector_type(8)))  float    v8f;

#define DD0  19
#define DD1  9
#define DD2  4
#define HH   200
#define WW   176
#define HWSZ (HH * WW)          // 35200
#define CINN 128
#define CC   64
#define NVOX 40000
#define BNEPS 1e-3f

union AFrag { v16h v; v8h h[2]; };

// ---------------------------------------------------------------------------
// utility kernels
// ---------------------------------------------------------------------------
__global__ void zero16_kernel(uint4* __restrict__ p, long long n) {
    long long i = (long long)blockIdx.x * blockDim.x + threadIdx.x;
    const long long stride = (long long)gridDim.x * blockDim.x;
    const uint4 z = make_uint4(0u, 0u, 0u, 0u);
    for (; i < n; i += stride) p[i] = z;
}

// wf[t][co][ci] = w[co][ci][t]   (OIDHW with kernel dims flattened to ntap)
__global__ void reform_w_kernel(const float* __restrict__ w, _Float16* __restrict__ wf,
                                int cin, int ntap) {
    const int idx = blockIdx.x * 256 + threadIdx.x;
    const int total = ntap * 64 * cin;
    if (idx >= total) return;
    const int t   = idx / (64 * cin);
    const int rem = idx - t * 64 * cin;
    const int co  = rem / cin;
    const int ci  = rem - co * cin;
    wf[idx] = (_Float16)w[(long)(co * cin + ci) * ntap + t];
}

// scatter-add voxel features into dense f16 grid, set occupancy mask
__global__ void scatter_kernel(const float* __restrict__ vf, const int* __restrict__ coors,
                               _Float16* __restrict__ dense, float* __restrict__ mask0) {
    const int idx = blockIdx.x * 256 + threadIdx.x;
    if (idx >= NVOX * 64) return;
    const int nv = idx >> 6;
    const int cp = (idx & 63) << 1;
    const int z = coors[nv * 4 + 1];
    const int y = coors[nv * 4 + 2];
    const int x = coors[nv * 4 + 3];
    const long p = (long)(z * HH + y) * WW + x;
    union { _Float16 h[2]; unsigned int u; } pk;
    pk.h[0] = (_Float16)vf[(long)nv * CINN + cp];
    pk.h[1] = (_Float16)vf[(long)nv * CINN + cp + 1];
    unsigned long long addr = (unsigned long long)(dense + p * CINN + cp);
    // CDNA5 packed f16 atomic add (duplicate voxels accumulate like the reference)
    asm volatile("global_atomic_pk_add_f16 %0, %1, off" :: "v"(addr), "v"(pk.u) : "memory");
    if (cp == 0) mask0[p] = 1.0f;
}

// mask_out[z] = (mask_in[2z] + mask_in[2z+1] + mask_in[2z+2]) > 0
__global__ void mask_down_kernel(const float* __restrict__ mi, float* __restrict__ mo, int Dout) {
    const int idx = blockIdx.x * 256 + threadIdx.x;
    if (idx >= Dout * HWSZ) return;
    const int z = idx / HWSZ;
    const int p = idx - z * HWSZ;
    const float s = mi[(2 * z) * HWSZ + p] + mi[(2 * z + 1) * HWSZ + p] + mi[(2 * z + 2) * HWSZ + p];
    mo[idx] = (s > 0.f) ? 1.f : 0.f;
}

// ---------------------------------------------------------------------------
// shared epilogue: (optional pre-mask) -> BN -> ReLU -> mask -> f16 store
// ---------------------------------------------------------------------------
__device__ __forceinline__ void conv_epilogue(
    const v8f acc0[4], const v8f acc1[4], int m, int hi8,
    int z, int y0, int x0,
    const float* __restrict__ mask,
    const float* __restrict__ bng, const float* __restrict__ bnb,
    const float* __restrict__ bnm, const float* __restrict__ bnv,
    int bnidx, _Float16* __restrict__ out)
{
#pragma unroll
    for (int t = 0; t < 4; ++t) {
        const int ch = t * 16 + m;
        const float sc = bng[bnidx * 64 + ch] * rsqrtf(bnv[bnidx * 64 + ch] + BNEPS);
        const float me = bnm[bnidx * 64 + ch];
        const float be = bnb[bnidx * 64 + ch];
#pragma unroll
        for (int rr = 0; rr < 2; ++rr) {
            const v8f* acc = rr ? acc1 : acc0;
            const long zy = (long)(z * HH + (y0 + rr)) * WW;
#pragma unroll
            for (int r = 0; r < 8; ++r) {
                const long p = zy + x0 + r + hi8;
                const float mk = mask[p];
                float v = acc[t][r] * mk;        // pre-mask (conv*mask) then BN
                v = (v - me) * sc + be;
                v = fmaxf(v, 0.f) * mk;
                out[p * 64 + ch] = (_Float16)v;
            }
        }
    }
}

// ---------------------------------------------------------------------------
// FAST 3x3x3 conv (stride 1, pad 1): interior tiles only.
// x-tiles 1..9 (all lanes in-bounds for dx in {-1,0,1}),
// y-pairs with y0 in [2,196] (both rows in-bounds for dy in {-1,0,1}).
// Fully branchless body; dx/kc/t displacements fold into load offsets.
// ---------------------------------------------------------------------------
template<int CINT>
__global__ __launch_bounds__(32) void conv3_wmma_fast(
    const _Float16* __restrict__ in, int Din,
    const _Float16* __restrict__ wt,             // [27][64][CINT] f16
    const float* __restrict__ mask,
    const float* __restrict__ bng, const float* __restrict__ bnb,
    const float* __restrict__ bnm, const float* __restrict__ bnv,
    int bnidx, _Float16* __restrict__ out)
{
    const int lane = threadIdx.x;
    const int m    = lane & 15;
    const int hi   = lane >> 4;
    const int hi8  = hi * 8;
    const int x0   = (1 + blockIdx.x) * 16;      // x-tiles 1..9
    const int y0   = 2 + blockIdx.y * 2;         // rows y0, y0+1 in [2,197]
    const int z    = blockIdx.z;
    const int x    = x0 + m;

    const v8f fz = {0.f,0.f,0.f,0.f,0.f,0.f,0.f,0.f};
    v8f acc0[4] = {fz, fz, fz, fz};
    v8f acc1[4] = {fz, fz, fz, fz};

    for (int dz = -1; dz <= 1; ++dz) {
        const int zi = z + dz;
        if (zi < 0 || zi >= Din) continue;
        const _Float16* inz = in + (long)zi * HWSZ * CINT;
        const _Float16* wz  = wt + (long)((dz + 1) * 9) * 64 * CINT;
#pragma unroll
        for (int dy = -1; dy <= 1; ++dy) {
            const _Float16* a0p = inz + ((long)(y0 + dy) * WW + x) * CINT;
            const _Float16* a1p = a0p + (long)WW * CINT;
            const _Float16* wb  = wz + ((long)((dy + 1) * 3) * 64 + m) * CINT + hi * 16;
#pragma unroll
            for (int dx = -1; dx <= 1; ++dx) {
#pragma unroll
                for (int kc = 0; kc < CINT; kc += 32) {
                    AFrag a0, a1;
                    a0.h[0] = *reinterpret_cast<const v8h*>(a0p + dx * CINT + kc + hi * 16);
                    a0.h[1] = *reinterpret_cast<const v8h*>(a0p + dx * CINT + kc + 8 + hi * 16);
                    a1.h[0] = *reinterpret_cast<const v8h*>(a1p + dx * CINT + kc + hi * 16);
                    a1.h[1] = *reinterpret_cast<const v8h*>(a1p + dx * CINT + kc + 8 + hi * 16);
                    v16h b[4];
#pragma unroll
                    for (int t = 0; t < 4; ++t)
                        b[t] = *reinterpret_cast<const v16h*>(
                            wb + (long)(dx + 1) * 64 * CINT + t * 16 * CINT + kc);
#pragma unroll
                    for (int t = 0; t < 4; ++t) {
                        acc0[t] = __builtin_amdgcn_wmma_f32_16x16x32_f16(
                            false, a0.v, false, b[t], (short)0, acc0[t], false, false);
                        acc1[t] = __builtin_amdgcn_wmma_f32_16x16x32_f16(
                            false, a1.v, false, b[t], (short)0, acc1[t], false, false);
                    }
                }
            }
        }
    }

    conv_epilogue(acc0, acc1, m, hi8, z, y0, x0, mask, bng, bnb, bnm, bnv, bnidx, out);
}

// ---------------------------------------------------------------------------
// EDGE 3x3x3 conv: boundary tiles (xt in {0,10} or y0 in {0,198}).
// Full per-lane x checks and per-row y checks; interior blocks exit early.
// ---------------------------------------------------------------------------
template<int CINT>
__global__ __launch_bounds__(32) void conv3_wmma_edge(
    const _Float16* __restrict__ in, int Din,
    const _Float16* __restrict__ wt,             // [27][64][CINT] f16
    const float* __restrict__ mask,
    const float* __restrict__ bng, const float* __restrict__ bnb,
    const float* __restrict__ bnm, const float* __restrict__ bnv,
    int bnidx, _Float16* __restrict__ out)
{
    const int xt = blockIdx.x;                   // 0..10
    const int y0 = blockIdx.y * 2;               // 0..198
    if (xt >= 1 && xt <= 9 && y0 >= 2 && y0 <= 196) return;  // fast kernel covers

    const int lane = threadIdx.x;
    const int m    = lane & 15;
    const int hi   = lane >> 4;
    const int hi8  = hi * 8;
    const int x0   = xt * 16;
    const int z    = blockIdx.z;
    const int x    = x0 + m;

    const v8f fz = {0.f,0.f,0.f,0.f,0.f,0.f,0.f,0.f};
    v8f acc0[4] = {fz, fz, fz, fz};
    v8f acc1[4] = {fz, fz, fz, fz};
    const v8h hz = {0,0,0,0,0,0,0,0};

    for (int dz = -1; dz <= 1; ++dz) {
        const int zi = z + dz;
        if (zi < 0 || zi >= Din) continue;
        for (int dy = -1; dy <= 1; ++dy) {
            const int yi0 = y0 + dy;
            const int yi1 = y0 + 1 + dy;
            const bool rv0 = (yi0 >= 0) && (yi0 < HH);
            const bool rv1 = (yi1 >= 0) && (yi1 < HH);
            if (!rv0 && !rv1) continue;
            for (int dx = -1; dx <= 1; ++dx) {
                const int xi = x + dx;
                const bool av = (xi >= 0) && (xi < WW);
                const _Float16* ap0 = in + ((long)(zi * HH + yi0) * WW + xi) * CINT;
                const _Float16* ap1 = in + ((long)(zi * HH + yi1) * WW + xi) * CINT;
                const int tap = (dz + 1) * 9 + (dy + 1) * 3 + (dx + 1);
                const _Float16* wr0 = wt + (long)(tap * 64 + m) * CINT + hi * 16;
#pragma unroll
                for (int kc = 0; kc < CINT; kc += 32) {
                    AFrag a0, a1;
                    if (rv0 && av) {
                        a0.h[0] = *reinterpret_cast<const v8h*>(ap0 + kc + hi * 16);
                        a0.h[1] = *reinterpret_cast<const v8h*>(ap0 + kc + 8 + hi * 16);
                    } else { a0.h[0] = hz; a0.h[1] = hz; }
                    if (rv1 && av) {
                        a1.h[0] = *reinterpret_cast<const v8h*>(ap1 + kc + hi * 16);
                        a1.h[1] = *reinterpret_cast<const v8h*>(ap1 + kc + 8 + hi * 16);
                    } else { a1.h[0] = hz; a1.h[1] = hz; }
                    v16h b[4];
#pragma unroll
                    for (int t = 0; t < 4; ++t)
                        b[t] = *reinterpret_cast<const v16h*>(wr0 + t * 16 * CINT + kc);
#pragma unroll
                    for (int t = 0; t < 4; ++t) {
                        acc0[t] = __builtin_amdgcn_wmma_f32_16x16x32_f16(
                            false, a0.v, false, b[t], (short)0, acc0[t], false, false);
                        acc1[t] = __builtin_amdgcn_wmma_f32_16x16x32_f16(
                            false, a1.v, false, b[t], (short)0, acc1[t], false, false);
                    }
                }
            }
        }
    }

    conv_epilogue(acc0, acc1, m, hi8, z, y0, x0, mask, bng, bnb, bnm, bnv, bnidx, out);
}

// ---------------------------------------------------------------------------
// 3x1x1 conv, stride (2,1,1), VALID. K = 3*64 = 192.
// TO_FLOAT=false: f16 channels-last out (down1). TO_FLOAT=true: final f32
// output with the (C, D, H, W)->(C*D, H, W) reshape channel mapping (down2).
// ---------------------------------------------------------------------------
template<bool TO_FLOAT>
__global__ __launch_bounds__(32) void downz_wmma(
    const _Float16* __restrict__ in,             // [Din][H][W][64]
    const _Float16* __restrict__ wt,             // [3][64][64] f16
    const float* __restrict__ mask,              // output-resolution mask
    const float* __restrict__ bng, const float* __restrict__ bnb,
    const float* __restrict__ bnm, const float* __restrict__ bnv,
    int bnidx, _Float16* __restrict__ outh, float* __restrict__ outf)
{
    const int lane = threadIdx.x;
    const int m    = lane & 15;
    const int hi   = lane >> 4;
    const int hi8  = hi * 8;
    const int x0   = blockIdx.x * 16;
    const int y    = blockIdx.y;
    const int z    = blockIdx.z;                 // output depth index
    const int x    = x0 + m;

    const v8f fz = {0.f,0.f,0.f,0.f,0.f,0.f,0.f,0.f};
    v8f acc[4] = {fz, fz, fz, fz};

    const _Float16* ap0 = in + ((long)((2 * z) * HH + y) * WW + x) * 64;
    const _Float16* wb  = wt + (long)m * 64 + hi * 16;
#pragma unroll
    for (int dz = 0; dz < 3; ++dz) {
#pragma unroll
        for (int kc = 0; kc < 64; kc += 32) {
            AFrag a;
            a.h[0] = *reinterpret_cast<const v8h*>(ap0 + (long)dz * HWSZ * 64 + kc + hi * 16);
            a.h[1] = *reinterpret_cast<const v8h*>(ap0 + (long)dz * HWSZ * 64 + kc + 8 + hi * 16);
            v16h b[4];
#pragma unroll
            for (int t = 0; t < 4; ++t)
                b[t] = *reinterpret_cast<const v16h*>(wb + (long)dz * 64 * 64 + t * 16 * 64 + kc);
#pragma unroll
            for (int t = 0; t < 4; ++t)
                acc[t] = __builtin_amdgcn_wmma_f32_16x16x32_f16(
                    false, a.v, false, b[t], (short)0, acc[t], false, false);
        }
    }

    const long zy = (long)(z * HH + y) * WW;
#pragma unroll
    for (int t = 0; t < 4; ++t) {
        const int ch = t * 16 + m;
        const float sc = bng[bnidx * 64 + ch] * rsqrtf(bnv[bnidx * 64 + ch] + BNEPS);
        const float me = bnm[bnidx * 64 + ch];
        const float be = bnb[bnidx * 64 + ch];
#pragma unroll
        for (int r = 0; r < 8; ++r) {
            const int mm = r + hi8;
            const long p = zy + x0 + mm;
            const float mk = mask[p];
            float v = acc[t][r];                  // no pre-mask for down convs
            v = (v - me) * sc + be;
            v = fmaxf(v, 0.f) * mk;
            if (TO_FLOAT) {
                outf[((long)(ch * DD2 + z) * HH + y) * WW + x0 + mm] = v;
            } else {
                outh[p * 64 + ch] = (_Float16)v;
            }
        }
    }
}

// ---------------------------------------------------------------------------
// workspace layout (bytes)
// ---------------------------------------------------------------------------
#define OFS_DENSE  0LL                                   // 171,212,800 (f16 19x200x176x128)
#define OFS_ACT1   171212800LL                           //  85,606,400 (f16 19x200x176x64)
#define OFS_MASK0  256819200LL                           //   2,675,200 (f32)
#define OFS_MASK1  259494400LL                           //   1,267,200 (f32)
#define OFS_MASK2  260761600LL                           //     563,200 (f32)
#define OFS_WF1    261324800LL                           //     442,368
#define OFS_WD1    261767168LL                           //      24,576
#define OFS_WF2    261791744LL                           //     221,184
#define OFS_WF3    262012928LL                           //     221,184
#define OFS_WD2    262234112LL                           //      24,576
// activations after down1 reuse the (dead) dense region:
#define OFS_ACT2   0LL                                   //  40,550,400 (f16 9x200x176x64)
#define OFS_ACT3   41943040LL                            //  40,550,400
#define OFS_ACT4   83886080LL                            //  40,550,400

extern "C" void kernel_launch(void* const* d_in, const int* in_sizes, int n_in,
                              void* d_out, int out_size, void* d_ws, size_t ws_size,
                              hipStream_t stream) {
    (void)in_sizes; (void)n_in; (void)out_size; (void)ws_size;

    const float* vf      = (const float*)d_in[0];
    const int*   coors   = (const int*)d_in[1];
    const float* w_sub1  = (const float*)d_in[3];
    const float* w_down1 = (const float*)d_in[4];
    const float* w_sub2  = (const float*)d_in[5];
    const float* w_sub3  = (const float*)d_in[6];
    const float* w_down2 = (const float*)d_in[7];
    const float* bng     = (const float*)d_in[8];
    const float* bnb     = (const float*)d_in[9];
    const float* bnm     = (const float*)d_in[10];
    const float* bnv     = (const float*)d_in[11];
    float* out = (float*)d_out;

    char* ws = (char*)d_ws;
    _Float16* dense = (_Float16*)(ws + OFS_DENSE);
    _Float16* act1  = (_Float16*)(ws + OFS_ACT1);
    _Float16* act2  = (_Float16*)(ws + OFS_ACT2);
    _Float16* act3  = (_Float16*)(ws + OFS_ACT3);
    _Float16* act4  = (_Float16*)(ws + OFS_ACT4);
    float* mask0 = (float*)(ws + OFS_MASK0);
    float* mask1 = (float*)(ws + OFS_MASK1);
    float* mask2 = (float*)(ws + OFS_MASK2);
    _Float16* wf1 = (_Float16*)(ws + OFS_WF1);
    _Float16* wd1 = (_Float16*)(ws + OFS_WD1);
    _Float16* wf2 = (_Float16*)(ws + OFS_WF2);
    _Float16* wf3 = (_Float16*)(ws + OFS_WF3);
    _Float16* wd2 = (_Float16*)(ws + OFS_WD2);

    // 1) zero dense grid + occupancy mask (every call: harness poisons ws once)
    zero16_kernel<<<4096, 256, 0, stream>>>((uint4*)dense, 171212800LL / 16);
    zero16_kernel<<<512,  256, 0, stream>>>((uint4*)mask0, 2675200LL / 16);

    // 2) weight reformat to [tap][cout][cin] f16
    reform_w_kernel<<<(27 * 64 * CINN + 255) / 256, 256, 0, stream>>>(w_sub1,  wf1, CINN, 27);
    reform_w_kernel<<<(3  * 64 * 64   + 255) / 256, 256, 0, stream>>>(w_down1, wd1, 64, 3);
    reform_w_kernel<<<(27 * 64 * 64   + 255) / 256, 256, 0, stream>>>(w_sub2,  wf2, 64, 27);
    reform_w_kernel<<<(27 * 64 * 64   + 255) / 256, 256, 0, stream>>>(w_sub3,  wf3, 64, 27);
    reform_w_kernel<<<(3  * 64 * 64   + 255) / 256, 256, 0, stream>>>(w_down2, wd2, 64, 3);

    // 3) scatter voxels (packed f16 atomic add) + mask set
    scatter_kernel<<<(NVOX * 64 + 255) / 256, 256, 0, stream>>>(vf, coors, dense, mask0);

    // 4) conv1: 128->64, 3x3x3, bn0, mask0   (interior + edge)
    conv3_wmma_fast<CINN><<<dim3(9, 98, DD0), 32, 0, stream>>>(
        dense, DD0, wf1, mask0, bng, bnb, bnm, bnv, 0, act1);
    conv3_wmma_edge<CINN><<<dim3(11, 100, DD0), 32, 0, stream>>>(
        dense, DD0, wf1, mask0, bng, bnb, bnm, bnv, 0, act1);

    // 5) mask downsample 19->9, then down1: bn1, mask1
    mask_down_kernel<<<(DD1 * HWSZ + 255) / 256, 256, 0, stream>>>(mask0, mask1, DD1);
    downz_wmma<false><<<dim3(WW / 16, HH, DD1), 32, 0, stream>>>(
        act1, wd1, mask1, bng, bnb, bnm, bnv, 1, act2, nullptr);

    // 6) conv2: 64->64, 3x3x3, bn2, mask1
    conv3_wmma_fast<64><<<dim3(9, 98, DD1), 32, 0, stream>>>(
        act2, DD1, wf2, mask1, bng, bnb, bnm, bnv, 2, act3);
    conv3_wmma_edge<64><<<dim3(11, 100, DD1), 32, 0, stream>>>(
        act2, DD1, wf2, mask1, bng, bnb, bnm, bnv, 2, act3);

    // 7) conv3: 64->64, 3x3x3, bn3, mask1
    conv3_wmma_fast<64><<<dim3(9, 98, DD1), 32, 0, stream>>>(
        act3, DD1, wf3, mask1, bng, bnb, bnm, bnv, 3, act4);
    conv3_wmma_edge<64><<<dim3(11, 100, DD1), 32, 0, stream>>>(
        act3, DD1, wf3, mask1, bng, bnb, bnm, bnv, 3, act4);

    // 8) mask downsample 9->4, then down2: bn4, mask2 -> f32 output (C*D,H,W)
    mask_down_kernel<<<(DD2 * HWSZ + 255) / 256, 256, 0, stream>>>(mask1, mask2, DD2);
    downz_wmma<true><<<dim3(WW / 16, HH, DD2), 32, 0, stream>>>(
        act4, wd2, mask2, bng, bnb, bnm, bnv, 4, nullptr, out);
}
